// VectorMerge_20942260535358
// MI455X (gfx1250) — matrix operand order repaired and checked by reading, not compiled, not run on hardware
//
#include <hip/hip_runtime.h>
#include <hip/hip_bf16.h>

// ---------------------------------------------------------------------------
// Fused 3-stream LN -> ReLU -> (out GEMM + gate GEMM) -> softmax-gate merge
// for MI455X (gfx1250, wave32, WGP, WMMA).
//
// Shapes: B=32768 rows; streams d = {1024, 512, 2048}; OUT=1024; gate=3*OUT.
// ~0.96 TFLOP over ~0.67 GB unique bytes -> HBM-bound (29 us floor at 23.3
// TB/s). Strategy: single fused kernel, activations cached as bf16 in the
// 320KB/WGP LDS, weights (59 MB) stay L2-resident, math via
// v_wmma_f32_16x16x32_bf16 with f32 accumulation. All global traffic uses
// b128 loads; LDS staging uses packed b32/b64 stores.
// ---------------------------------------------------------------------------

typedef __attribute__((ext_vector_type(16))) __bf16 v16bf;
typedef __attribute__((ext_vector_type(8)))  float  v8f;

#define BROWS   32768
#define OUT     1024
#define D_A     1024
#define D_B     512
#define D_C     2048
#define DSUM    (D_A + D_B + D_C)   // 3584
#define TM      16                  // rows per block (one WMMA M tile)
#define TN      128                 // cols per col-tile (8 waves x 16)
#define KC      32                  // K chunk (WMMA bf16 K)
#define LN_EPS  1e-5f

// LDS layout (ushort units):
//   fbuf : TM * DSUM               = 57344   (bf16 activations, all 3 streams)
//   wbuf : KC * TN                 = 4096    (W tile, transposed [n][k])
//   gbuf : 3 * KC * TN             = 12288   (Wg tiles, transposed [n][k])
// total 73728 ushorts = 147456 bytes of dynamic LDS (needs >64KB -> CDNA5 WGP)
#define FBUF_US   (TM * DSUM)
#define WBUF_US   (KC * TN)
#define SMEM_US   (FBUF_US + 4 * WBUF_US)
#define SMEM_BYTES (SMEM_US * 2)

__device__ __forceinline__ unsigned short f2bf(float f) {
  unsigned u = __float_as_uint(f);
  unsigned r = (u + 0x7FFFu + ((u >> 16) & 1u)) >> 16;  // round-to-nearest-even
  return (unsigned short)r;
}

// pack two floats into two bf16 in one dword: low = a, high = b
__device__ __forceinline__ unsigned pk2bf(float a, float b) {
  return (unsigned)f2bf(a) | ((unsigned)f2bf(b) << 16);
}

union FragU {
  uint4 q[2];
  v16bf v;
};

__global__ void __launch_bounds__(256)
fused_ln_gemm_gate_kernel(
    const float* __restrict__ x0,  const float* __restrict__ lw0, const float* __restrict__ lb0,
    const float* __restrict__ W0,  const float* __restrict__ b0,
    const float* __restrict__ G0,  const float* __restrict__ bg0,
    const float* __restrict__ x1,  const float* __restrict__ lw1, const float* __restrict__ lb1,
    const float* __restrict__ W1,  const float* __restrict__ b1,
    const float* __restrict__ G1,  const float* __restrict__ bg1,
    const float* __restrict__ x2,  const float* __restrict__ lw2, const float* __restrict__ lb2,
    const float* __restrict__ W2,  const float* __restrict__ b2,
    const float* __restrict__ G2,  const float* __restrict__ bg2,
    float* __restrict__ out)
{
  extern __shared__ unsigned short smem[];
  unsigned short* fbuf = smem;                       // activations (bf16)
  unsigned short* wbuf = smem + FBUF_US;             // W tile  [n][k]
  unsigned short* gbuf = wbuf + WBUF_US;             // Wg tiles [i][n][k]

  const int tid  = (int)threadIdx.x;
  const int lane = tid & 31;                         // wave32
  const int wave = tid >> 5;                         // 8 waves
  const int rowBase = (int)blockIdx.x * TM;

  const int Dd[3] = {D_A, D_B, D_C};
  const unsigned fOff[3] = {0u, (unsigned)(TM * D_A), (unsigned)(TM * (D_A + D_B))};
  const float* xs[3]  = {x0, x1, x2};
  const float* lws[3] = {lw0, lw1, lw2};
  const float* lbs[3] = {lb0, lb1, lb2};
  const float* Ws[3]  = {W0, W1, W2};
  const float* bs[3]  = {b0, b1, b2};
  const float* Gs[3]  = {G0, G1, G2};
  const float* bgs[3] = {bg0, bg1, bg2};

  // ---------------- Phase 1: LayerNorm + ReLU -> bf16 into LDS -------------
  // 8 waves x 2 rows each = 16 rows per stream. float4 loads (b128),
  // wave-level shfl reductions (warpSize == 32), packed 8B LDS stores.
#pragma unroll
  for (int s = 0; s < 3; ++s) {
    const int d = Dd[s];
    const float* __restrict__ x  = xs[s];
    const float* __restrict__ lw = lws[s];
    const float* __restrict__ lb = lbs[s];
#pragma unroll
    for (int rr = 0; rr < 2; ++rr) {
      const int r = wave * 2 + rr;
      const float* __restrict__ xr = x + (size_t)(rowBase + r) * (size_t)d;
      float sum = 0.f, sq = 0.f;
      for (int i = lane * 4; i < d; i += 128) {
        float4 v = *(const float4*)(xr + i);
        sum += v.x + v.y + v.z + v.w;
        sq  += v.x * v.x + v.y * v.y + v.z * v.z + v.w * v.w;
      }
#pragma unroll
      for (int off = 16; off > 0; off >>= 1) {
        sum += __shfl_xor(sum, off, 32);
        sq  += __shfl_xor(sq,  off, 32);
      }
      const float inv_d = 1.f / (float)d;
      const float mu  = sum * inv_d;
      const float var = sq * inv_d - mu * mu;
      const float rs  = rsqrtf(var + LN_EPS);
      unsigned short* frow = fbuf + fOff[s] + (unsigned)r * (unsigned)d;
      for (int i = lane * 4; i < d; i += 128) {
        float4 v = *(const float4*)(xr + i);
        float4 w = *(const float4*)(lw + i);
        float4 c = *(const float4*)(lb + i);
        ushort4 o;
        o.x = f2bf(fmaxf((v.x - mu) * rs * w.x + c.x, 0.f));
        o.y = f2bf(fmaxf((v.y - mu) * rs * w.y + c.y, 0.f));
        o.z = f2bf(fmaxf((v.z - mu) * rs * w.z + c.z, 0.f));
        o.w = f2bf(fmaxf((v.w - mu) * rs * w.w + c.w, 0.f));
        *(ushort4*)(frow + i) = o;   // 8-byte DS store
      }
    }
  }
  __syncthreads();

  // ---------------- Phase 2: WMMA GEMMs + softmax gate merge ----------------
  const int nLocal = wave * 16 + (lane & 15);   // column within 128-col tile
  const int kSel   = lane >> 4;                 // 0: lanes 0-15, 1: lanes 16-31

  for (int ct = 0; ct < OUT / TN; ++ct) {
    const int colBase = ct * TN;
    const int col = colBase + nLocal;

    v8f accO[3], accG[3];
#pragma unroll
    for (int i = 0; i < 3; ++i) {
      const float bo = bs[i][col];
      const float bg = bgs[0][i * OUT + col] + bgs[1][i * OUT + col] + bgs[2][i * OUT + col];
#pragma unroll
      for (int e = 0; e < 8; ++e) { accO[i][e] = bo; accG[i][e] = bg; }
    }

#pragma unroll
    for (int s = 0; s < 3; ++s) {
      const int d = Dd[s];
      const float* __restrict__ W = Ws[s];
      const float* __restrict__ G = Gs[s];
      const unsigned short* fb = fbuf + fOff[s];

      for (int k0 = 0; k0 < d; k0 += KC) {
        __syncthreads();  // protect staging buffers from previous iteration

        // Cooperative stage of W / Wg K-chunk, f32 -> bf16, transposed [n][k].
        // Each thread: 4 columns (float4 / b128 global loads) x 2 consecutive
        // K rows, so each transposed write is a packed dword (ds_store_b32).
        // KC/2 * TN/4 = 512 units over 256 threads -> 2 iterations.
        for (int q = tid; q < (KC / 2) * (TN / 4); q += 256) {
          const int k  = (q >> 5) * 2;       // 0,2,..,30
          const int n4 = (q & 31) * 4;       // 0,4,..,124
          {
            const float* __restrict__ wr = W + (size_t)(k0 + k) * OUT + colBase + n4;
            float4 r0 = *(const float4*)(wr);
            float4 r1 = *(const float4*)(wr + OUT);
            const float* a0 = (const float*)&r0;
            const float* a1 = (const float*)&r1;
#pragma unroll
            for (int j = 0; j < 4; ++j)
              *(unsigned*)(wbuf + (n4 + j) * KC + k) = pk2bf(a0[j], a1[j]);
          }
          const float* __restrict__ gr = G + (size_t)(k0 + k) * (size_t)(3 * OUT) + colBase + n4;
#pragma unroll
          for (int i = 0; i < 3; ++i) {
            float4 r0 = *(const float4*)(gr + i * OUT);
            float4 r1 = *(const float4*)(gr + i * OUT + 3 * OUT);
            const float* a0 = (const float*)&r0;
            const float* a1 = (const float*)&r1;
#pragma unroll
            for (int j = 0; j < 4; ++j)
              *(unsigned*)(gbuf + i * WBUF_US + (n4 + j) * KC + k) = pk2bf(a0[j], a1[j]);
          }
        }
        // Speculative prefetch of next weight chunk (global_prefetch_b8)
        if (k0 + KC < d) {
          __builtin_prefetch(&W[(size_t)(k0 + KC) * OUT + colBase + (tid & (TN - 1))], 0, 1);
          __builtin_prefetch(&G[(size_t)(k0 + KC) * (size_t)(3 * OUT) + colBase + (tid & (TN - 1))], 0, 1);
        }
        __syncthreads();

        // A fragment (16x32, contiguous k order chosen here, not the ISA table order): row M = lane%16,
        // 16 consecutive K starting at kSel*16 within the chunk (matches B).
        FragU fa;
        const unsigned short* ap = fb + (unsigned)(lane & 15) * (unsigned)d + (unsigned)k0 + (unsigned)(kSel * 16);
        fa.q[0] = *(const uint4*)(ap);
        fa.q[1] = *(const uint4*)(ap + 8);

        // B fragment (32x16, contiguous k order chosen here, same as A): col N = lane%16,
        // 16 consecutive K starting at kSel*16 -> contiguous in [n][k] LDS.
        FragU fw;
        const unsigned short* wp = wbuf + (unsigned)(nLocal * KC) + (unsigned)(kSel * 16);
        fw.q[0] = *(const uint4*)(wp);
        fw.q[1] = *(const uint4*)(wp + 8);

        accO[s] = __builtin_amdgcn_wmma_f32_16x16x32_bf16(
            false, fa.v, false, fw.v, (short)0, accO[s], false, false);

#pragma unroll
        for (int i = 0; i < 3; ++i) {
          FragU fg;
          const unsigned short* gp = gbuf + (unsigned)(i * WBUF_US + nLocal * KC) + (unsigned)(kSel * 16);
          fg.q[0] = *(const uint4*)(gp);
          fg.q[1] = *(const uint4*)(gp + 8);
          accG[i] = __builtin_amdgcn_wmma_f32_16x16x32_bf16(
              false, fa.v, false, fg.v, (short)0, accG[i], false, false);
        }
      }
    }

    // Softmax over the 3 streams (elementwise across accumulators) + merge.
    v8f res;
#pragma unroll
    for (int e = 0; e < 8; ++e) {
      const float g0 = accG[0][e], g1 = accG[1][e], g2 = accG[2][e];
      const float mx = fmaxf(g0, fmaxf(g1, g2));
      const float e0 = __expf(g0 - mx);
      const float e1 = __expf(g1 - mx);
      const float e2 = __expf(g2 - mx);
      const float inv = 1.f / (e0 + e1 + e2);
      res[e] = (e0 * accO[0][e] + e1 * accO[1][e] + e2 * accO[2][e]) * inv;
    }

    // C/D 16x16 f32 layout: VGPR e -> M = e + (lane>=16 ? 8 : 0), N = lane%16.
    const int mBase = kSel * 8;
    float* __restrict__ orow = out + (size_t)(rowBase + mBase) * OUT + colBase + wave * 16 + (lane & 15);
#pragma unroll
    for (int e = 0; e < 8; ++e) {
      orow[(size_t)e * OUT] = res[e];
    }
  }
}

extern "C" void kernel_launch(void* const* d_in, const int* in_sizes, int n_in,
                              void* d_out, int out_size, void* d_ws, size_t ws_size,
                              hipStream_t stream) {
  (void)in_sizes; (void)n_in; (void)d_ws; (void)ws_size; (void)out_size;

  // setup_inputs() order: for each stream {a,b,c}: x, ln_w, ln_b, W, b, Wg, bg
  const float* x0  = (const float*)d_in[0];
  const float* lw0 = (const float*)d_in[1];
  const float* lb0 = (const float*)d_in[2];
  const float* W0  = (const float*)d_in[3];
  const float* b0  = (const float*)d_in[4];
  const float* G0  = (const float*)d_in[5];
  const float* bg0 = (const float*)d_in[6];
  const float* x1  = (const float*)d_in[7];
  const float* lw1 = (const float*)d_in[8];
  const float* lb1 = (const float*)d_in[9];
  const float* W1  = (const float*)d_in[10];
  const float* b1  = (const float*)d_in[11];
  const float* G1  = (const float*)d_in[12];
  const float* bg1 = (const float*)d_in[13];
  const float* x2  = (const float*)d_in[14];
  const float* lw2 = (const float*)d_in[15];
  const float* lb2 = (const float*)d_in[16];
  const float* W2  = (const float*)d_in[17];
  const float* b2  = (const float*)d_in[18];
  const float* G2  = (const float*)d_in[19];
  const float* bg2 = (const float*)d_in[20];
  float* out = (float*)d_out;

  // Allow >64KB dynamic LDS (CDNA5 WGP has 320KB). Ignore failure.
  (void)hipFuncSetAttribute((const void*)fused_ln_gemm_gate_kernel,
                            hipFuncAttributeMaxDynamicSharedMemorySize,
                            SMEM_BYTES);

  dim3 grid(BROWS / TM);   // 2048 blocks
  dim3 block(256);         // 8 wave32 waves
  fused_ln_gemm_gate_kernel<<<grid, block, SMEM_BYTES, stream>>>(
      x0, lw0, lb0, W0, b0, G0, bg0,
      x1, lw1, lb1, W1, b1, G1, bg1,
      x2, lw2, lb2, W2, b2, G2, bg2,
      out);
}
